// Decomposed_GSA_18880676233302
// MI455X (gfx1250) — matrix operand order repaired and checked by reading, not compiled, not run on hardware
//
#include <hip/hip_runtime.h>

#define BB  16
#define HH  64
#define WW  64
#define CC  256
#define NHD 8
#define KDM 32
#define MTOT (BB*HH*WW)   // 65536 pixels

typedef __attribute__((ext_vector_type(16))) _Float16 v16h;
typedef __attribute__((ext_vector_type(8)))  float    v8f;
typedef __attribute__((ext_vector_type(4)))  unsigned int v4u;

union FragU { v16h h; v4u u[2]; };

// A-fragment (16x32 f16): lane L holds row m=L%16; halves 0..15 -> K = 2*b0..2*b0+15
// contiguous, b0 = (L/16)*8.  Two 16B loads.
__device__ __forceinline__ v16h ldA(const _Float16* row, int b0) {
  FragU f;
  f.u[0] = *reinterpret_cast<const v4u*>(row + 2 * b0);
  f.u[1] = *reinterpret_cast<const v4u*>(row + 2 * b0 + 8);
  return f.h;
}
// B-fragment (32x16 f16): lane L holds col n=L%16; K = (L/16)*16 + 0..15
// contiguous.  p must already point at col-base + (L/16)*16.
__device__ __forceinline__ v16h ldB(const _Float16* p) {
  FragU f;
  f.u[0] = *reinterpret_cast<const v4u*>(p);
  f.u[1] = *reinterpret_cast<const v4u*>(p + 8);
  return f.h;
}
__device__ __forceinline__ v8f wmma32(v16h a, v16h b, v8f c) {
  return __builtin_amdgcn_wmma_f32_16x16x32_f16(false, a, false, b,
                                                (short)0, c, false, false);
}

// ---------------------------------------------------------------- converts
__global__ __launch_bounds__(256) void cvt_f32_f16(const float* __restrict__ src,
                                                   _Float16* __restrict__ dst, int n) {
  int i = blockIdx.x * 256 + threadIdx.x;
  if (i < n) dst[i] = (_Float16)src[i];
}

// W is (K=256, N=256) row-major; store N-major f16 so B-fragments are contiguous.
__global__ __launch_bounds__(256) void cvt_w_t(const float* __restrict__ W,
                                               _Float16* __restrict__ Wt) {
  int i = blockIdx.x * 256 + threadIdx.x;       // 65536
  int k = i >> 8, n = i & 255;
  Wt[(size_t)n * CC + k] = (_Float16)W[i];
}

// ---------------------------------------------------------------- WMMA GEMM
// Y[m][n] = sum_k X[m][k] * Wt[n][k] + bias[n];  M=65536, N=256, K=256.
// 256 threads = 8 waves / block; each wave computes a 32(M) x 64(N) strip with
// a double-buffered K loop so WMMA only waits on the previous chunk's loads.
__global__ __launch_bounds__(256) void gemm_f16(const _Float16* __restrict__ X,
                                                const _Float16* __restrict__ Wt,
                                                const float*    __restrict__ bias,
                                                _Float16* __restrict__ outH,
                                                float*    __restrict__ outF) {
  const int wave = (blockIdx.x << 3) | (threadIdx.x >> 5);
  const int lane = threadIdx.x & 31;
  const int lm = lane & 15, lh = lane >> 4;
  const int nG = wave & 3;          // 4 column groups of 64
  const int mT = wave >> 2;         // 2048 row tiles of 32
  const _Float16* xrow0 = X + (size_t)((mT << 5) + lm) * CC;
  const _Float16* xrow1 = xrow0 + (size_t)16 * CC;
  const _Float16* wbase = Wt + (size_t)((nG << 6) + lm) * CC + lh * 16;

  v16h a[2][2], bf[2][4];
  a[0][0] = ldA(xrow0, lh << 3);
  a[0][1] = ldA(xrow1, lh << 3);
  #pragma unroll
  for (int t = 0; t < 4; ++t) bf[0][t] = ldB(wbase + (size_t)(t << 4) * CC);

  v8f c[2][4] = {};
  #pragma unroll
  for (int kk = 0; kk < 8; ++kk) {
    const int cur = kk & 1, nxt = cur ^ 1;
    if (kk < 7) {                           // prefetch next K chunk
      const int ko = (kk + 1) << 5;
      a[nxt][0] = ldA(xrow0 + ko, lh << 3);
      a[nxt][1] = ldA(xrow1 + ko, lh << 3);
      #pragma unroll
      for (int t = 0; t < 4; ++t)
        bf[nxt][t] = ldB(wbase + ko + (size_t)(t << 4) * CC);
    }
    #pragma unroll
    for (int t = 0; t < 4; ++t) {
      c[0][t] = wmma32(a[cur][0], bf[cur][t], c[0][t]);
      c[1][t] = wmma32(a[cur][1], bf[cur][t], c[1][t]);
    }
  }
  #pragma unroll
  for (int rt = 0; rt < 2; ++rt) {
    const int mBase = (mT << 5) + (rt << 4) + (lh << 3);
    #pragma unroll
    for (int t = 0; t < 4; ++t) {
      const int n = (nG << 6) + (t << 4) + lm;
      const float bv = bias[n];
      #pragma unroll
      for (int r = 0; r < 8; ++r) {
        float v = c[rt][t][r] + bv;
        size_t idx = (size_t)(mBase + r) * CC + n;
        if (outH) outH[idx] = (_Float16)v; else outF[idx] = v;
      }
    }
  }
}

// ---------------------------------------------------------------- rotary (in place)
__global__ __launch_bounds__(256) void rotary(_Float16* __restrict__ q,
                                              _Float16* __restrict__ k,
                                              const float* __restrict__ sn,
                                              const float* __restrict__ cs) {
  const int t  = blockIdx.x * 256 + threadIdx.x;   // pair index, 8388608 total
  const int pc = t & 127;
  const int pix = t >> 7;
  const int w = pix & 63;
  const int h = (pix >> 6) & 63;
  const int c0 = pc << 1;
  const int d0 = c0 & (KDM - 1);
  const size_t sidx = (size_t)(h * WW + w) * KDM + d0;
  const float s0 = sn[sidx], s1 = sn[sidx + 1];
  const float cA = cs[sidx], cB = cs[sidx + 1];
  const size_t qi = (size_t)pix * CC + c0;
  float q0 = (float)q[qi], q1 = (float)q[qi + 1];
  q[qi]     = (_Float16)(q0 * cA - q1 * s0);
  q[qi + 1] = (_Float16)(q1 * cB + q0 * s1);
  const float sc = 0.17677669529663687f;           // KD^-0.5
  float k0 = (float)k[qi] * sc, k1 = (float)k[qi + 1] * sc;
  k[qi]     = (_Float16)(k0 * cA - k1 * s0);
  k[qi + 1] = (_Float16)(k1 * cB + k0 * s1);
}

// ---------------------------------------------------------------- axis attention
// One wave32 owns one (b, pos, head) instance: S = Q(64x32) Kt(32x64) + mask,
// softmax over rows, O = P(64x64) V(64x32).  MODE 0: along W, MODE 1: along H.
template <int MODE>
__global__ __launch_bounds__(128) void attn_axis(const _Float16* __restrict__ Q,
                                                 const _Float16* __restrict__ K,
                                                 const _Float16* __restrict__ V,
                                                 const float*    __restrict__ mask,
                                                 _Float16* __restrict__ outH) {
  __shared__ __align__(16) _Float16 lds[4 * 6144];   // per wave: P 64x64 + Vt 32x64
  const int wid  = threadIdx.x >> 5;
  const int lane = threadIdx.x & 31;
  const int lm = lane & 15, lh = lane >> 4;
  const int inst = (blockIdx.x << 2) | wid;          // 8192 instances
  const int n = inst & (NHD - 1);
  const int p = (inst >> 3) & 63;                    // h (MODE 0) or w (MODE 1)
  const int b = inst >> 9;

  size_t qoff, qstride, voff, vstride, moff;
  if (MODE == 0) {
    qoff = (size_t)(b * HH + p) * WW * CC + n * KDM;  qstride = CC;
    voff = qoff;                                      vstride = CC;
    moff = (size_t)(n * HH + p) * WW * WW;
  } else {
    qoff = ((size_t)b * HH * WW + p) * CC + n * KDM;  qstride = (size_t)WW * CC;
    voff = (size_t)((b * WW + p) * NHD + n) * HH * KDM; vstride = KDM;
    moff = (size_t)(n * WW + p) * HH * HH;
  }
  const _Float16* qb = Q + qoff;
  const _Float16* kb = K + qoff;
  const _Float16* vb = V + voff;
  const float*    mb = mask + moff;
  _Float16* ldsP  = lds + wid * 6144;
  _Float16* ldsVt = ldsP + 4096;

  // hoist all Q/K fragments so the 16 QK^T WMMAs issue as one dense burst
  v16h af[4], bf[4];
  #pragma unroll
  for (int tt = 0; tt < 4; ++tt) {
    af[tt] = ldA(qb + (size_t)(tt * 16 + lm) * qstride, lh << 3);
    bf[tt] = ldB(kb + (size_t)(tt * 16 + lm) * qstride + lh * 16);
  }

  // stage V transposed into LDS: Vt[d][j] (independent; overlaps WMMA below)
  #pragma unroll
  for (int jj = 0; jj < 2; ++jj) {
    const int j = lane + jj * 32;
    const _Float16* vp = vb + (size_t)j * vstride;
    union { v4u u[4]; _Float16 h[32]; } vr;
    vr.u[0] = *reinterpret_cast<const v4u*>(vp);
    vr.u[1] = *reinterpret_cast<const v4u*>(vp + 8);
    vr.u[2] = *reinterpret_cast<const v4u*>(vp + 16);
    vr.u[3] = *reinterpret_cast<const v4u*>(vp + 24);
    #pragma unroll
    for (int d = 0; d < 32; ++d) ldsVt[d * 64 + j] = vr.h[d];
  }

  // S = Q Kt  (16 wmma)
  v8f s[4][4] = {};
  #pragma unroll
  for (int ti = 0; ti < 4; ++ti)
    #pragma unroll
    for (int tj = 0; tj < 4; ++tj)
      s[ti][tj] = wmma32(af[ti], bf[tj], s[ti][tj]);

  // mask + row softmax (rows split 4 elems/lane + 16-lane shfl reduce), P -> LDS
  #pragma unroll
  for (int ti = 0; ti < 4; ++ti) {
    #pragma unroll
    for (int r = 0; r < 8; ++r) {
      const int row = ti * 16 + (lh << 3) + r;
      const float* mrow = mb + (size_t)row * 64;
      float e[4];
      float mx = -1e30f;
      #pragma unroll
      for (int tj = 0; tj < 4; ++tj) {
        e[tj] = s[ti][tj][r] + mrow[tj * 16 + lm];
        mx = fmaxf(mx, e[tj]);
      }
      mx = fmaxf(mx, __shfl_xor(mx, 1));
      mx = fmaxf(mx, __shfl_xor(mx, 2));
      mx = fmaxf(mx, __shfl_xor(mx, 4));
      mx = fmaxf(mx, __shfl_xor(mx, 8));
      float sum = 0.f;
      #pragma unroll
      for (int tj = 0; tj < 4; ++tj) { e[tj] = __expf(e[tj] - mx); sum += e[tj]; }
      sum += __shfl_xor(sum, 1);
      sum += __shfl_xor(sum, 2);
      sum += __shfl_xor(sum, 4);
      sum += __shfl_xor(sum, 8);
      const float inv = 1.0f / sum;
      #pragma unroll
      for (int tj = 0; tj < 4; ++tj)
        ldsP[row * 64 + tj * 16 + lm] = (_Float16)(e[tj] * inv);
    }
  }
  // same-wave DS ops are in order: no barrier needed before re-reading LDS.

  // O = P V  (16 wmma, K=64 in two chunks)
  #pragma unroll
  for (int ti = 0; ti < 4; ++ti) {
    v8f o[2] = {};
    #pragma unroll
    for (int kc = 0; kc < 2; ++kc) {
      v16h ap = ldA(ldsP + (ti * 16 + lm) * 64 + kc * 32, lh << 3);
      #pragma unroll
      for (int dt = 0; dt < 2; ++dt) {
        v16h bv = ldB(ldsVt + (dt * 16 + lm) * 64 + kc * 32 + lh * 16);
        o[dt] = wmma32(ap, bv, o[dt]);
      }
    }
    #pragma unroll
    for (int dt = 0; dt < 2; ++dt) {
      #pragma unroll
      for (int r = 0; r < 8; ++r) {
        const int i = ti * 16 + (lh << 3) + r;
        const int d = dt * 16 + lm;
        size_t idx;
        if (MODE == 0)   // vmid layout [b][w=i][n][h=p][d]
          idx = ((((size_t)b * WW + i) * NHD + n) * HH + p) * KDM + d;
        else             // attn output [b][h=i][w=p][c]
          idx = (((size_t)b * HH + i) * WW + p) * CC + n * KDM + d;
        outH[idx] = (_Float16)o[dt][r];
      }
    }
  }
}

// ---------------------------------------------------------------- LEPE 5x5 dwconv, add in place
__global__ __launch_bounds__(256) void lepe_add(const _Float16* __restrict__ Vh,
                                                const float* __restrict__ dwk,
                                                const float* __restrict__ dwb,
                                                _Float16* __restrict__ attn) {
  const size_t idx = (size_t)blockIdx.x * 256 + threadIdx.x;
  const int c = (int)(idx & 255);
  const int pix = (int)(idx >> 8);
  const int w = pix & 63;
  const int h = (pix >> 6) & 63;
  const int b = pix >> 12;
  float acc = dwb[c];
  #pragma unroll
  for (int dy = 0; dy < 5; ++dy) {
    const int hy = h + dy - 2;
    if ((unsigned)hy < 64u) {
      #pragma unroll
      for (int dx = 0; dx < 5; ++dx) {
        const int wx = w + dx - 2;
        if ((unsigned)wx < 64u)
          acc += (float)Vh[(((size_t)b * HH + hy) * WW + wx) * CC + c] *
                 dwk[(dy * 5 + dx) * CC + c];
      }
    }
  }
  attn[idx] = (_Float16)((float)attn[idx] + acc);
}

// ---------------------------------------------------------------- launch
extern "C" void kernel_launch(void* const* d_in, const int* in_sizes, int n_in,
                              void* d_out, int out_size, void* d_ws, size_t ws_size,
                              hipStream_t stream) {
  const float* x      = (const float*)d_in[0];
  const float* sn     = (const float*)d_in[1];
  const float* cs     = (const float*)d_in[2];
  const float* mask_h = (const float*)d_in[3];
  const float* mask_w = (const float*)d_in[4];
  const float* Wq = (const float*)d_in[5];
  const float* bq = (const float*)d_in[6];
  const float* Wk = (const float*)d_in[7];
  const float* bk = (const float*)d_in[8];
  const float* Wv = (const float*)d_in[9];
  const float* bv = (const float*)d_in[10];
  const float* dwk = (const float*)d_in[11];
  const float* dwb = (const float*)d_in[12];
  const float* Wo = (const float*)d_in[13];
  const float* bo = (const float*)d_in[14];
  float* out = (float*)d_out;

  char* ws = (char*)d_ws;
  size_t off = 0;
  auto alloc = [&](size_t bytes) {
    char* p = ws + off;
    off += (bytes + 255) & ~(size_t)255;
    return p;
  };
  const size_t BIG = (size_t)MTOT * CC * sizeof(_Float16);   // 32 MB
  _Float16* x_h  = (_Float16*)alloc(BIG);
  _Float16* wq_t = (_Float16*)alloc((size_t)CC * CC * 2);
  _Float16* wk_t = (_Float16*)alloc((size_t)CC * CC * 2);
  _Float16* wv_t = (_Float16*)alloc((size_t)CC * CC * 2);
  _Float16* wo_t = (_Float16*)alloc((size_t)CC * CC * 2);
  _Float16* q_h  = (_Float16*)alloc(BIG);
  _Float16* k_h  = (_Float16*)alloc(BIG);
  _Float16* v_h  = (_Float16*)alloc(BIG);
  _Float16* vmid = (_Float16*)alloc(BIG);
  _Float16* ao_h = (_Float16*)alloc(BIG);

  cvt_f32_f16<<<65536, 256, 0, stream>>>(x, x_h, MTOT * CC);
  cvt_w_t<<<256, 256, 0, stream>>>(Wq, wq_t);
  cvt_w_t<<<256, 256, 0, stream>>>(Wk, wk_t);
  cvt_w_t<<<256, 256, 0, stream>>>(Wv, wv_t);
  cvt_w_t<<<256, 256, 0, stream>>>(Wo, wo_t);

  gemm_f16<<<1024, 256, 0, stream>>>(x_h, wq_t, bq, q_h, nullptr);
  gemm_f16<<<1024, 256, 0, stream>>>(x_h, wk_t, bk, k_h, nullptr);
  gemm_f16<<<1024, 256, 0, stream>>>(x_h, wv_t, bv, v_h, nullptr);

  rotary<<<32768, 256, 0, stream>>>(q_h, k_h, sn, cs);

  attn_axis<0><<<2048, 128, 0, stream>>>(q_h, k_h, v_h, mask_w, vmid);
  attn_axis<1><<<2048, 128, 0, stream>>>(q_h, k_h, vmid, mask_h, ao_h);

  lepe_add<<<65536, 256, 0, stream>>>(v_h, dwk, dwb, ao_h);

  gemm_f16<<<1024, 256, 0, stream>>>(ao_h, wo_t, bo, nullptr, out);
}
